// Attention_7739531067877
// MI455X (gfx1250) — matrix operand order repaired and checked by reading, not compiled, not run on hardware
//
#include <hip/hip_runtime.h>
#include <hip/hip_bf16.h>

// ---------------------------------------------------------------------------
// Multi-head self-attention forward for MI455X (gfx1250), bf16 WMMA + f32 acc.
// N=4096, DIM=1024, HEADS=16, DHEAD=64.
// ---------------------------------------------------------------------------

typedef __attribute__((ext_vector_type(16))) __bf16 v16bf;
typedef __attribute__((ext_vector_type(8)))  __bf16 v8bf;
typedef __attribute__((ext_vector_type(8)))  float  v8f;
typedef __attribute__((ext_vector_type(4)))  int    v4i;

#define N_TOK  4096
#define DIM    1024
#define HEADS  16
#define DHEAD  64
#define INNER  1024
#define NQKV   3072   // Q|K|V concatenated output columns

// ---- WMMA fragment loaders (16x16x32 bf16, contiguous k order used by this code, wave32) ----------
// A (16xK=32): lane L holds row M=L&15; elements 0..15 = K [16*half .. +15]
//              (half = L>>4).
__device__ __forceinline__ v16bf load_a_frag(const __bf16* base, int ld,
                                             int lane) {
  const int m = lane & 15, half = lane >> 4;
  const __bf16* r = base + (size_t)m * ld + 16 * half;
  union { v16bf v; v8bf h[2]; } u;
  u.h[0] = *(const v8bf*)(r);
  u.h[1] = *(const v8bf*)(r + 8);
  return u.v;
}

// B (K=32 x 16), loaded from B^T stored row-major [N][K]:
// lane L holds column N=L&15, K elements 16*half + e (contiguous 32B).
__device__ __forceinline__ v16bf load_b_frag(const __bf16* baseT, int ld,
                                             int lane) {
  const int n = lane & 15, half = lane >> 4;
  return *(const v16bf*)(baseT + (size_t)n * ld + 16 * half);
}

__device__ __forceinline__ v8f wmma_bf16(v16bf a, v16bf b, v8f c) {
  return __builtin_amdgcn_wmma_f32_16x16x32_bf16(false, a, false, b,
                                                 (short)0, c, false, false);
}

// ---- CDNA5 async global->LDS copy (16B per lane), with safe fallback ------
__device__ __forceinline__ void async_cp16(const __bf16* g, __bf16* l) {
#if __has_builtin(__builtin_amdgcn_global_load_async_to_lds_b128)
  __builtin_amdgcn_global_load_async_to_lds_b128(
      (__attribute__((address_space(1))) v4i*)g,
      (__attribute__((address_space(3))) v4i*)l, 0, 0);
#else
  *(v8bf*)l = *(const v8bf*)g;          // sync fallback: load + ds_store
#endif
}

__device__ __forceinline__ void wait_async_lds() {
#if __has_builtin(__builtin_amdgcn_global_load_async_to_lds_b128)
#if __has_builtin(__builtin_amdgcn_s_wait_asynccnt)
  __builtin_amdgcn_s_wait_asynccnt(0);
#else
  asm volatile("s_wait_asynccnt 0" ::: "memory");
#endif
#endif
}

// ---- Kernel 1a: x f32 -> bf16 ---------------------------------------------
__global__ void k_cvt_x(const float* __restrict__ x, __bf16* __restrict__ xb,
                        int n) {
  int i = blockIdx.x * blockDim.x + threadIdx.x;
  if (i < n) xb[i] = (__bf16)x[i];
}

// ---- Kernel 1b: Wq|Wk|Wv (k-major) -> Wt[3072][1024] (n-major, bf16) ------
__global__ void k_build_wt(const float* __restrict__ Wq,
                           const float* __restrict__ Wk,
                           const float* __restrict__ Wv,
                           __bf16* __restrict__ Wt) {
  int tid = blockIdx.x * blockDim.x + threadIdx.x;      // 0 .. 3*1024*1024-1
  int sel = tid >> 20;
  int rem = tid & ((1 << 20) - 1);
  int k = rem >> 10, inner = rem & 1023;
  const float* W = (sel == 0) ? Wq : ((sel == 1) ? Wk : Wv);
  Wt[((size_t)(sel << 10 | inner) << 10) | k] = (__bf16)W[rem];
}

// ---- Kernel 1c: Wo[j][d] -> Wot[d][j] bf16 --------------------------------
__global__ void k_build_wot(const float* __restrict__ Wo,
                            __bf16* __restrict__ Wot) {
  int tid = blockIdx.x * blockDim.x + threadIdx.x;      // 0 .. 1024*1024-1
  int j = tid >> 10, d = tid & 1023;
  Wot[((size_t)d << 10) | j] = (__bf16)Wo[tid];
}

// ---- ping-pong GEMM core: 16(M) x 64(N) strip, K in steps of 64 -----------
// Two statically-named fragment sets -> no register rotation copies; loads
// for one set overlap the 4-WMMA burst of the other set.
__device__ __forceinline__ void gemm_strip(const __bf16* pA, const __bf16* pB,
                                           int K, int lane, v8f acc[4]) {
  v16bf a0 = load_a_frag(pA, K, lane);
  v16bf b00 = load_b_frag(pB + (size_t)0  * K, K, lane);
  v16bf b01 = load_b_frag(pB + (size_t)16 * K, K, lane);
  v16bf b02 = load_b_frag(pB + (size_t)32 * K, K, lane);
  v16bf b03 = load_b_frag(pB + (size_t)48 * K, K, lane);

  for (int k0 = 0; k0 < K; k0 += 64) {
    // fetch set1 (k0+32); K is a multiple of 64 so always in range
    v16bf a1  = load_a_frag(pA + k0 + 32, K, lane);
    v16bf b10 = load_b_frag(pB + (size_t)0  * K + k0 + 32, K, lane);
    v16bf b11 = load_b_frag(pB + (size_t)16 * K + k0 + 32, K, lane);
    v16bf b12 = load_b_frag(pB + (size_t)32 * K + k0 + 32, K, lane);
    v16bf b13 = load_b_frag(pB + (size_t)48 * K + k0 + 32, K, lane);

    acc[0] = wmma_bf16(a0, b00, acc[0]);
    acc[1] = wmma_bf16(a0, b01, acc[1]);
    acc[2] = wmma_bf16(a0, b02, acc[2]);
    acc[3] = wmma_bf16(a0, b03, acc[3]);

    if (k0 + 64 < K) {                   // fetch set0 for next iteration
      a0  = load_a_frag(pA + k0 + 64, K, lane);
      b00 = load_b_frag(pB + (size_t)0  * K + k0 + 64, K, lane);
      b01 = load_b_frag(pB + (size_t)16 * K + k0 + 64, K, lane);
      b02 = load_b_frag(pB + (size_t)32 * K + k0 + 64, K, lane);
      b03 = load_b_frag(pB + (size_t)48 * K + k0 + 64, K, lane);
    }

    acc[0] = wmma_bf16(a1, b10, acc[0]);
    acc[1] = wmma_bf16(a1, b11, acc[1]);
    acc[2] = wmma_bf16(a1, b12, acc[2]);
    acc[3] = wmma_bf16(a1, b13, acc[3]);
  }
}

// ---- Kernel 2: fused QKV GEMM ---------------------------------------------
// grid (N/16, 3072/512), block 256 (8 waves). Wave -> 16x64 output strip.
__global__ void __launch_bounds__(256)
k_qkv(const __bf16* __restrict__ xb, const __bf16* __restrict__ Wt,
      __bf16* __restrict__ Qb, __bf16* __restrict__ Kb,
      __bf16* __restrict__ Vt) {
  const int lane = threadIdx.x & 31;
  const int wave = threadIdx.x >> 5;
  const int m0 = blockIdx.x * 16;
  const int n0 = blockIdx.y * 512 + wave * 64;

  v8f acc[4] = {};
  gemm_strip(xb + (size_t)m0 * DIM, Wt + (size_t)n0 * DIM, DIM, lane, acc);

  const int nl = lane & 15, half = lane >> 4;
#pragma unroll
  for (int t = 0; t < 4; ++t) {
    int j = n0 + 16 * t + nl;              // global output column in [0,3072)
    int sel = j >> 10, inner = j & 1023;
    int head = inner >> 6, d = inner & 63;
    if (sel < 2) {                          // Q or K: [head][m][d]
      __bf16* dst = sel ? Kb : Qb;
#pragma unroll
      for (int r = 0; r < 8; ++r) {
        int m = m0 + r + 8 * half;
        dst[((size_t)(head * N_TOK + m) << 6) | d] = (__bf16)acc[t][r];
      }
    } else {                                // V transposed: [head][d][m]
      v8bf pk;
#pragma unroll
      for (int r = 0; r < 8; ++r) pk[r] = (__bf16)acc[t][r];
      *(v8bf*)(Vt + (size_t)(head * DHEAD + d) * N_TOK + m0 + 8 * half) = pk;
    }
  }
}

// ---- Kernel 3: flash attention (transposed S / O) -------------------------
// grid (N/64, HEADS), block 128 (4 waves). Wave owns 16 query columns.
// K/V tiles double-buffered in LDS via async global->LDS copies; all four
// waves consume the shared tiles.  S^T = K x Q^T ; O^T = V^T x P^T.
__global__ void __launch_bounds__(128)
k_attn(const __bf16* __restrict__ Qb, const __bf16* __restrict__ Kb,
       const __bf16* __restrict__ Vt, __bf16* __restrict__ Ob) {
  __shared__ __bf16 Ks[2][32 * DHEAD];     // 2 x 4KB  (32 keys x 64 dims)
  __shared__ __bf16 Vs[2][DHEAD * 32];     // 2 x 4KB  (64 dims x 32 keys)

  const int tid = threadIdx.x;
  const int lane = tid & 31;
  const int wave = tid >> 5;
  const int half = lane >> 4;
  const int head = blockIdx.y;
  const int q0 = blockIdx.x * 64 + wave * 16;

  const __bf16* Qh = Qb + (size_t)head * N_TOK * DHEAD;
  const __bf16* Kh = Kb + (size_t)head * N_TOK * DHEAD;
  const __bf16* Vh = Vt + (size_t)head * DHEAD * N_TOK;

  // Stage one 32-key block into LDS buffer `buf` (128 threads x 4 x 16B).
  auto stage = [&](int buf, int kb) {
#pragma unroll
    for (int i = 0; i < 2; ++i) {          // K block: 4KB contiguous
      int c = tid + 128 * i;               // chunk of 8 bf16
      async_cp16(Kh + (size_t)kb * DHEAD + c * 8, &Ks[buf][c * 8]);
    }
#pragma unroll
    for (int i = 0; i < 2; ++i) {          // V^T block: 64 rows x 32 cols
      int c = tid + 128 * i;
      int row = c >> 2, off = (c & 3) * 8;
      async_cp16(Vh + (size_t)row * N_TOK + kb + off, &Vs[buf][row * 32 + off]);
    }
  };

  // Q^T B-fragments (K-chunks d0 = 0, 32), loaded once from global.
  const v16bf bq0 = load_b_frag(Qh + (size_t)q0 * DHEAD + 0, DHEAD, lane);
  const v16bf bq1 = load_b_frag(Qh + (size_t)q0 * DHEAD + 32, DHEAD, lane);

  v8f o[4] = {};                    // O^T accum: 4 d-tiles of 16, f32
  float m_run = -__builtin_inff();
  float l_run = 0.0f;
  const float SCALE = 0.125f;       // 1/sqrt(64)
  const float LOG2E = 1.44269504088896340736f;

  stage(0, 0);

  for (int kb = 0; kb < N_TOK; kb += 32) {
    const int buf = (kb >> 5) & 1;
    wait_async_lds();               // this wave's staged tiles are in LDS
    __syncthreads();                // everyone's tiles visible; prev buf free

    if (kb + 32 < N_TOK) stage(buf ^ 1, kb + 32);   // overlap next block
    if (kb + 64 < N_TOK)                            // deep L2 prefetch
      __builtin_prefetch(Kh + (size_t)(kb + 64 + lane) * DHEAD, 0, 1);

    // S^T tiles: rows = keys (kb+16t .. +15), cols = queries.
    v8f st[2] = {};
#pragma unroll
    for (int t = 0; t < 2; ++t) {
      v16bf ak0 = load_a_frag(&Ks[buf][(16 * t) * DHEAD + 0], DHEAD, lane);
      st[t] = wmma_bf16(ak0, bq0, st[t]);
      v16bf ak1 = load_a_frag(&Ks[buf][(16 * t) * DHEAD + 32], DHEAD, lane);
      st[t] = wmma_bf16(ak1, bq1, st[t]);
    }

    // ---- online softmax over keys (per query column = per lane) ----
    float ts0[8], ts1[8];
    float cur = -__builtin_inff();
#pragma unroll
    for (int r = 0; r < 8; ++r) {
      ts0[r] = st[0][r] * SCALE;
      ts1[r] = st[1][r] * SCALE;
      cur = fmaxf(cur, fmaxf(ts0[r], ts1[r]));
    }
    cur = fmaxf(cur, __shfl_xor(cur, 16, 32));       // combine key halves
    const float m_new = fmaxf(m_run, cur);
    const float alpha = exp2f((m_run - m_new) * LOG2E);

    float p0[8], p1[8], rs = 0.0f;
#pragma unroll
    for (int r = 0; r < 8; ++r) {
      p0[r] = exp2f((ts0[r] - m_new) * LOG2E);
      p1[r] = exp2f((ts1[r] - m_new) * LOG2E);
      rs += p0[r] + p1[r];
    }
    rs += __shfl_xor(rs, 16, 32);
    l_run = l_run * alpha + rs;
    m_run = m_new;

#pragma unroll
    for (int dt = 0; dt < 4; ++dt)
#pragma unroll
      for (int r = 0; r < 8; ++r) o[dt][r] *= alpha;

    // ---- build P^T B-fragment from S^T D-tiles via lane-half swap ----
    // B needs (per lane): col m = lane&15, keys 16*half + e, e=0..15.
    v16bf bp;
#pragma unroll
    for (int e = 0; e < 8; ++e) {
      float sw1 = __shfl_xor(p1[e], 16, 32);   // tile1 rows from partner half
      bp[e] = (__bf16)(half ? sw1 : p0[e]);
    }
#pragma unroll
    for (int e = 0; e < 8; ++e) {
      float sw0 = __shfl_xor(p0[e], 16, 32);   // tile0 rows from partner half
      bp[8 + e] = (__bf16)(half ? p1[e] : sw0);
    }

    // ---- O^T += V^T x P^T ----
#pragma unroll
    for (int dt = 0; dt < 4; ++dt) {
      v16bf av = load_a_frag(&Vs[buf][(16 * dt) * 32], 32, lane);
      o[dt] = wmma_bf16(av, bp, o[dt]);
    }
  }

  // normalize and store O as [n][head*64+d] bf16 (rows of D-tile are
  // contiguous in d -> one packed 16B store per d-tile per lane).
  const float inv = 1.0f / l_run;
  const int q = q0 + (lane & 15);
#pragma unroll
  for (int dt = 0; dt < 4; ++dt) {
    v8bf pk;
#pragma unroll
    for (int r = 0; r < 8; ++r) pk[r] = (__bf16)(o[dt][r] * inv);
    *(v8bf*)(Ob + (size_t)q * INNER + head * DHEAD + 16 * dt + 8 * half) = pk;
  }
}

// ---- Kernel 4: output projection out = Ob @ Wo + bo (f32 out) -------------
__global__ void __launch_bounds__(256)
k_proj(const __bf16* __restrict__ Ob, const __bf16* __restrict__ Wot,
       const float* __restrict__ bo, float* __restrict__ out) {
  const int lane = threadIdx.x & 31;
  const int wave = threadIdx.x >> 5;
  const int m0 = blockIdx.x * 16;
  const int n0 = blockIdx.y * 512 + wave * 64;

  v8f acc[4] = {};
  gemm_strip(Ob + (size_t)m0 * INNER, Wot + (size_t)n0 * INNER, INNER, lane,
             acc);

  const int nl = lane & 15, half = lane >> 4;
#pragma unroll
  for (int t = 0; t < 4; ++t) {
    int j = n0 + 16 * t + nl;
    float bias = bo[j];
#pragma unroll
    for (int r = 0; r < 8; ++r) {
      int m = m0 + r + 8 * half;
      out[(size_t)m * DIM + j] = acc[t][r] + bias;
    }
  }
}

// ---------------------------------------------------------------------------
extern "C" void kernel_launch(void* const* d_in, const int* in_sizes, int n_in,
                              void* d_out, int out_size, void* d_ws,
                              size_t ws_size, hipStream_t stream) {
  const float* x  = (const float*)d_in[0];
  const float* Wq = (const float*)d_in[1];
  const float* Wk = (const float*)d_in[2];
  const float* Wv = (const float*)d_in[3];
  const float* Wo = (const float*)d_in[4];
  const float* bo = (const float*)d_in[5];
  float* out = (float*)d_out;

  char* ws = (char*)d_ws;                 // 48 MB layout
  __bf16* xb  = (__bf16*)(ws + (size_t)0);
  __bf16* Wt  = (__bf16*)(ws + ((size_t)8  << 20));   // 3072x1024 bf16 (6MB)
  __bf16* Wot = (__bf16*)(ws + ((size_t)14 << 20));   // 1024x1024 bf16 (2MB)
  __bf16* Qb  = (__bf16*)(ws + ((size_t)16 << 20));   // [16][4096][64] (8MB)
  __bf16* Kb  = (__bf16*)(ws + ((size_t)24 << 20));   // [16][4096][64] (8MB)
  __bf16* Vt  = (__bf16*)(ws + ((size_t)32 << 20));   // [16][64][4096] (8MB)
  __bf16* Ob  = (__bf16*)(ws + ((size_t)40 << 20));   // [4096][1024]   (8MB)

  k_cvt_x   <<<(N_TOK * DIM) / 256, 256, 0, stream>>>(x, xb, N_TOK * DIM);
  k_build_wt<<<(3 * DIM * INNER) / 256, 256, 0, stream>>>(Wq, Wk, Wv, Wt);
  k_build_wot<<<(INNER * DIM) / 256, 256, 0, stream>>>(Wo, Wot);

  k_qkv <<<dim3(N_TOK / 16, NQKV / 512), 256, 0, stream>>>(xb, Wt, Qb, Kb, Vt);
  k_attn<<<dim3(N_TOK / 64, HEADS),      128, 0, stream>>>(Qb, Kb, Vt, Ob);
  k_proj<<<dim3(N_TOK / 16, DIM / 512),  256, 0, stream>>>(Ob, Wot, bo, out);
}
